// PointGNN_52862457479462
// MI455X (gfx1250) — matrix operand order repaired and checked by reading, not compiled, not run on hardware
//
#include <hip/hip_runtime.h>

// ---------------------------------------------------------------------------
// PointGNN forward for gfx1250 (MI455X). bf16 WMMA (16x16x32) everywhere;
// async global->LDS copies (ASYNCcnt) for activation gathers.
// wave32; one wave per workgroup; 16 rows per wave tile.
// ---------------------------------------------------------------------------

typedef __bf16 bf16_t;
typedef bf16_t v16bf __attribute__((ext_vector_type(16)));
typedef bf16_t v8bf  __attribute__((ext_vector_type(8)));
typedef float  v8f   __attribute__((ext_vector_type(8)));
typedef unsigned short u16;

// 16-byte payload type for the async global->LDS builtin (per hipcc diagnostic:
// param0 is 'int __attribute__((vector_size(16))) __device__ *').
typedef int v4i __attribute__((vector_size(4 * sizeof(int))));
typedef __attribute__((address_space(1))) v4i* gv4i_p;   // global (__device__)
typedef __attribute__((address_space(3))) v4i* lv4i_p;   // LDS (__shared__)

#define DEVINL __device__ __forceinline__

#if defined(__HIP_DEVICE_COMPILE__) && __has_builtin(__builtin_amdgcn_global_load_async_to_lds_b128)
#define HAS_ASYNC 1
#else
#define HAS_ASYNC 0
#endif

DEVINL u16 f2bfu(float f) { bf16_t h = (bf16_t)f; return __builtin_bit_cast(u16, h); }

// monotonic float<->uint encoding for atomic max over signed floats
DEVINL unsigned fenc(float f) {
    unsigned u = __float_as_uint(f);
    return (u & 0x80000000u) ? ~u : (u | 0x80000000u);
}
DEVINL float fdec(unsigned u) {
    u = (u & 0x80000000u) ? (u & 0x7fffffffu) : ~u;
    return __uint_as_float(u);
}
#define ENC_NEG_INF 0x007FFFFFu  // fenc(-inf)

// 16-byte async copy global(bf16) -> LDS; falls back to sync copy.
DEVINL void async16(const u16* g, bf16_t* l)
{
#if HAS_ASYNC
    __builtin_amdgcn_global_load_async_to_lds_b128(
        (gv4i_p)(void*)g, (lv4i_p)(void*)l, 0, 0);
#else
    *(v8bf*)l = *(const v8bf*)g;
#endif
}

DEVINL void async_wait()
{
#if HAS_ASYNC
#if __has_builtin(__builtin_amdgcn_s_wait_asynccnt)
    __builtin_amdgcn_s_wait_asynccnt(0);
#else
    asm volatile("s_wait_asynccnt 0x0" ::: "memory");
#endif
#endif
}

// Async-load 16 contiguous rows (320 bf16 each = 640B) from the bf16 mirror.
DEVINL void rows16_async(const u16* sbm, bf16_t* act, int r0, int lane)
{
    const u16* g = sbm + (size_t)r0 * 320;
    for (int idx = lane; idx < 16 * 40; idx += 32)   // 40 x 16B chunks per row
        async16(g + idx * 8, act + idx * 8);
    async_wait();
}

// ---------------------------------------------------------------------------
// Wave-level GEMM: out[16][Otiles*16] (f32, LDS) = act[16][Kpad] (bf16, LDS)
//                  x W[Opad][Kpad] (bf16, global, row=output col) + bias
// A-frag (16-bit A 16x32): lane m=l&15; g=l>>4; elems 0..7 -> K=kb+8g..,
// elems 8..15 -> K=kb+16+8g.. ; B-frag: lane col n=l&15, same K order as A.
// C layout: elem i of lane l -> row i+8*(l>>4), col nb*16+(l&15).
// ---------------------------------------------------------------------------
DEVINL void wave_gemm(const u16* __restrict__ Wu, const float* __restrict__ bp,
                      const bf16_t* act, float* outF,
                      int Kpad, int Otiles, int lane)
{
    const bf16_t* W = reinterpret_cast<const bf16_t*>(Wu);
    const int n = lane & 15, g = lane >> 4;
    const int Ost = Otiles * 16;
    for (int nb = 0; nb < Otiles; ++nb) {
        v8f acc = {0.f, 0.f, 0.f, 0.f, 0.f, 0.f, 0.f, 0.f};
        for (int kb = 0; kb < Kpad; kb += 32) {
            v8bf a0 = *(const v8bf*)(act + n * Kpad + kb + 8 * g);
            v8bf a1 = *(const v8bf*)(act + n * Kpad + kb + 16 + 8 * g);
            v16bf A;
#pragma unroll
            for (int i = 0; i < 8; ++i) { A[i] = a0[i]; A[i + 8] = a1[i]; }
            const bf16_t* wp = W + (size_t)(nb * 16 + n) * Kpad + kb + 8 * g;
            v8bf b0 = *(const v8bf*)wp;
            v8bf b1 = *(const v8bf*)(wp + 16);
            v16bf B;
#pragma unroll
            for (int i = 0; i < 8; ++i) { B[i] = b0[i]; B[i + 8] = b1[i]; }
            acc = __builtin_amdgcn_wmma_f32_16x16x32_bf16(
                false, A, false, B, (short)0, acc, false, false);
        }
        const int col = nb * 16 + n;
        const float bias = bp[col];
#pragma unroll
        for (int i = 0; i < 8; ++i) outF[(i + 8 * g) * Ost + col] = acc[i] + bias;
    }
}

// InstanceNorm(last dim, biased var, eps=1e-5) + ReLU over 16 rows in LDS.
DEVINL void wave_norm_relu(float* outF, bf16_t* nextAct,
                           int O, int Ostride, int KpadNext, int lane)
{
    const int row = lane & 15, half = lane >> 4;
    float s1 = 0.f, s2 = 0.f;
    for (int c = half; c < O; c += 2) {
        float v = outF[row * Ostride + c];
        s1 += v; s2 += v * v;
    }
    s1 += __shfl_xor(s1, 16, 32);
    s2 += __shfl_xor(s2, 16, 32);
    const float invO = 1.0f / (float)O;
    const float mean = s1 * invO;
    float var = s2 * invO - mean * mean;
    var = fmaxf(var, 0.0f);
    const float inv = rsqrtf(var + 1e-5f);
    for (int c = half; c < O; c += 2) {
        float v = fmaxf((outF[row * Ostride + c] - mean) * inv, 0.0f);
        outF[row * Ostride + c] = v;
        if (nextAct) nextAct[row * KpadNext + c] = (bf16_t)v;
    }
    if (nextAct)
        for (int c = O + half; c < KpadNext; c += 2)
            nextAct[row * KpadNext + c] = (bf16_t)0.f;
}

// ---------------------------------------------------------------------------
// Weight prep: w[I][O] f32 -> Wt[Opad][Kpad] bf16 (transposed, zero-padded),
// bias padded to Opad f32. perm=1 reorders edge-MLP input features so the
// gathered state vector sits first: new k in [0,300) -> old 3+k,
// [300,303) -> old k-300 (posdiff).
// ---------------------------------------------------------------------------
__global__ void k_prep(const float* __restrict__ w, const float* __restrict__ b,
                       u16* __restrict__ Wt, float* __restrict__ bp,
                       int I, int O, int Kpad, int perm)
{
    const int k = blockIdx.x * blockDim.x + threadIdx.x;
    const int o = blockIdx.y;
    if (k < Kpad) {
        int sk = k;
        if (perm) sk = (k < 300) ? (k + 3) : ((k < 303) ? (k - 300) : k);
        Wt[(size_t)o * Kpad + k] = f2bfu((sk < I && o < O) ? w[(size_t)sk * O + o] : 0.f);
    }
    if (k == 0) bp[o] = (o < O) ? b[o] : 0.f;
}

// Fuse lin2: Wf = w1 @ w2 (I x O), bf = b1 @ w2 + b2. Row i==I computes bias.
__global__ void k_fuse(const float* __restrict__ w1, const float* __restrict__ b1,
                       const float* __restrict__ w2, const float* __restrict__ b2,
                       float* __restrict__ Wf, float* __restrict__ bfo,
                       int I, int H, int O)
{
    const int o = blockIdx.x * blockDim.x + threadIdx.x;
    const int i = blockIdx.y;
    if (o >= O) return;
    float acc = 0.f;
    if (i < I) {
        for (int h = 0; h < H; ++h) acc += w1[(size_t)i * H + h] * w2[(size_t)h * O + o];
        Wf[(size_t)i * O + o] = acc;
    } else {
        for (int h = 0; h < H; ++h) acc += b1[h] * w2[(size_t)h * O + o];
        bfo[o] = acc + b2[o];
    }
}

__global__ void k_agg_init(unsigned* __restrict__ agg, int n)
{
    const int i = blockIdx.x * blockDim.x + threadIdx.x;
    if (i < n) agg[i] = ENC_NEG_INF;
}

// s (f32, [N][300]) -> bf16 mirror [N][320], zero-padded cols 300..319
__global__ void k_s2bf(const float* __restrict__ s, u16* __restrict__ sbm, int N)
{
    const int idx = blockIdx.x * blockDim.x + threadIdx.x;
    const int n = idx / 320, c = idx - n * 320;
    if (n < N) sbm[idx] = f2bfu(c < 300 ? s[(size_t)n * 300 + c] : 0.f);
}

// ---------------------------------------------------------------------------
// Init MLP: key_points [R][4] -> 4 blocks (Linear+INorm+ReLU) -> 300; then
// max over the 32 points of each node via int-bitwise atomicMax (values >= 0,
// s pre-zeroed).
// ---------------------------------------------------------------------------
__global__ void k_init(const float* __restrict__ kp,
                       const u16* W0, const float* b0, const u16* W1, const float* b1,
                       const u16* W2, const float* b2, const u16* W3, const float* b3,
                       float* __restrict__ s)
{
    __shared__ __align__(16) bf16_t act[16 * 128];
    __shared__ __align__(16) float  outF[16 * 304];
    const int lane = threadIdx.x;
    const int r0 = blockIdx.x * 16;
    for (int r = 0; r < 16; ++r)
        for (int c = lane; c < 32; c += 32)
            act[r * 32 + c] = (bf16_t)(c < 4 ? kp[(size_t)(r0 + r) * 4 + c] : 0.f);
    __syncthreads();
    wave_gemm(W0, b0, act, outF, 32, 2, lane);   __syncthreads();
    wave_norm_relu(outF, act, 32, 32, 32, lane); __syncthreads();
    wave_gemm(W1, b1, act, outF, 32, 4, lane);   __syncthreads();
    wave_norm_relu(outF, act, 64, 64, 64, lane); __syncthreads();
    wave_gemm(W2, b2, act, outF, 64, 8, lane);   __syncthreads();
    wave_norm_relu(outF, act, 128, 128, 128, lane); __syncthreads();
    wave_gemm(W3, b3, act, outF, 128, 19, lane); __syncthreads();
    wave_norm_relu(outF, nullptr, 300, 304, 0, lane); __syncthreads();
    const int row = lane & 15, half = lane >> 4;
    const int node = (r0 + row) >> 5;  // M_PTS = 32
    for (int c = half; c < 300; c += 2)
        atomicMax((int*)&s[(size_t)node * 300 + c], __float_as_int(outF[row * 304 + c]));
}

// dx = s @ Wh_fused + bh  (300 -> 3, stored padded to 16 cols)
__global__ void k_dx(const u16* __restrict__ sbm, const u16* Wh, const float* bh,
                     float* __restrict__ dxb)
{
    __shared__ __align__(16) bf16_t act[16 * 320];
    __shared__ __align__(16) float  outF[16 * 16];
    const int lane = threadIdx.x;
    const int r0 = blockIdx.x * 16;
    rows16_async(sbm, act, r0, lane);
    __syncthreads();
    wave_gemm(Wh, bh, act, outF, 320, 1, lane);
    __syncthreads();
    const int row = lane & 15, half = lane >> 4;
    for (int c = half; c < 16; c += 2)
        dxb[(size_t)(r0 + row) * 16 + c] = outF[row * 16 + c];
}

// Edge MLP (fused 303->300, features reordered to [s | posdiff | pad]) +
// segment_max via encoded-uint atomics. State rows gathered asynchronously
// from the bf16 mirror straight into LDS.
__global__ void k_edge(const u16* __restrict__ sbm, const float* __restrict__ pos,
                       const float* __restrict__ dxb,
                       const int* __restrict__ src, const int* __restrict__ dst,
                       const u16* Wf, const float* bf_,
                       unsigned* __restrict__ agg)
{
    __shared__ __align__(16) bf16_t act[16 * 320];
    __shared__ __align__(16) float  outF[16 * 304];
    __shared__ int sdst[16];
    __shared__ int ssrc[16];
    const int lane = threadIdx.x;
    const int e0 = blockIdx.x * 16;
    if (lane < 16) {
        const int e = e0 + lane;
        ssrc[lane] = src[e];
        sdst[lane] = dst[e];
    }
    __syncthreads();
    // async gather: full 640B rows (cols 300..319 of the mirror are zero)
    for (int idx = lane; idx < 16 * 40; idx += 32) {
        const int r = idx / 40, ch = idx - r * 40;
        async16(sbm + (size_t)ssrc[r] * 320 + ch * 8, act + r * 320 + ch * 8);
    }
    async_wait();
    __syncthreads();
    if (lane < 16) {
        const int js = ssrc[lane], jd = sdst[lane];
        for (int c = 0; c < 3; ++c)
            act[lane * 320 + 300 + c] =
                (bf16_t)(pos[(size_t)js * 3 + c] - pos[(size_t)jd * 3 + c] -
                         dxb[(size_t)jd * 16 + c]);
    }
    __syncthreads();
    wave_gemm(Wf, bf_, act, outF, 320, 19, lane);
    __syncthreads();
    const int row = lane & 15, half = lane >> 4;
    const int node = sdst[row];
    for (int c = half; c < 300; c += 2)
        atomicMax(&agg[(size_t)node * 300 + c], fenc(outF[row * 304 + c]));
}

// s += (where(isfinite(agg), agg, 0)) @ Wg_fused + bg ; refresh bf16 mirror
__global__ void k_update(float* __restrict__ s, u16* __restrict__ sbm,
                         const unsigned* __restrict__ agg,
                         const u16* Wg, const float* bg)
{
    __shared__ __align__(16) bf16_t act[16 * 320];
    __shared__ __align__(16) float  outF[16 * 304];
    const int lane = threadIdx.x;
    const int r0 = blockIdx.x * 16;
    for (int r = 0; r < 16; ++r)
        for (int c = lane; c < 320; c += 32) {
            float v = 0.f;
            if (c < 300) {
                const float f = fdec(agg[(size_t)(r0 + r) * 300 + c]);
                v = __builtin_isfinite(f) ? f : 0.f;
            }
            act[r * 320 + c] = (bf16_t)v;
        }
    __syncthreads();
    wave_gemm(Wg, bg, act, outF, 320, 19, lane);
    __syncthreads();
    const int row = lane & 15, half = lane >> 4;
    for (int c = half; c < 300; c += 2) {
        const float v = s[(size_t)(r0 + row) * 300 + c] + outF[row * 304 + c];
        s[(size_t)(r0 + row) * 300 + c] = v;
        sbm[(size_t)(r0 + row) * 320 + c] = f2bfu(v);
    }
}

// Classification + 3 localization heads fused in one kernel.
struct HeadW {
    const u16*   W[11];
    const float* B[11];
};

__global__ void k_heads(const u16* __restrict__ sbm, HeadW hw,
                        float* __restrict__ out, int N)
{
    __shared__ __align__(16) bf16_t sb[16 * 320];
    __shared__ __align__(16) bf16_t act[16 * 320];
    __shared__ __align__(16) float  outF[16 * 64];
    const int lane = threadIdx.x;
    const int r0 = blockIdx.x * 16;
    rows16_async(sbm, sb, r0, lane);
    __syncthreads();
    const int row = lane & 15, half = lane >> 4;

    // cls head: 300 -> 64 (IN+ReLU) -> 3 (IN+ReLU)
    for (int i = lane; i < 16 * 320; i += 32) act[i] = sb[i];
    __syncthreads();
    wave_gemm(hw.W[0], hw.B[0], act, outF, 320, 4, lane); __syncthreads();
    wave_norm_relu(outF, act, 64, 64, 64, lane);          __syncthreads();
    wave_gemm(hw.W[1], hw.B[1], act, outF, 64, 1, lane);  __syncthreads();
    wave_norm_relu(outF, nullptr, 3, 16, 0, lane);        __syncthreads();
    for (int c = half; c < 3; c += 2)
        out[(size_t)(r0 + row) * 3 + c] = outF[row * 16 + c];
    __syncthreads();

    // loc heads: 300 -> 64 -> 64 -> 7, each block IN+ReLU
    const size_t no3 = (size_t)N * 3;
    for (int ci = 0; ci < 3; ++ci) {
        for (int i = lane; i < 16 * 320; i += 32) act[i] = sb[i];
        __syncthreads();
        wave_gemm(hw.W[2 + 3 * ci], hw.B[2 + 3 * ci], act, outF, 320, 4, lane); __syncthreads();
        wave_norm_relu(outF, act, 64, 64, 64, lane);                            __syncthreads();
        wave_gemm(hw.W[3 + 3 * ci], hw.B[3 + 3 * ci], act, outF, 64, 4, lane);  __syncthreads();
        wave_norm_relu(outF, act, 64, 64, 64, lane);                            __syncthreads();
        wave_gemm(hw.W[4 + 3 * ci], hw.B[4 + 3 * ci], act, outF, 64, 1, lane);  __syncthreads();
        wave_norm_relu(outF, nullptr, 7, 16, 0, lane);                          __syncthreads();
        for (int c = half; c < 7; c += 2)
            out[no3 + (size_t)(r0 + row) * 21 + 7 * ci + c] = outF[row * 16 + c];
        __syncthreads();
    }
}

// ---------------------------------------------------------------------------
// Host side
// ---------------------------------------------------------------------------
extern "C" void kernel_launch(void* const* d_in, const int* in_sizes, int n_in,
                              void* d_out, int out_size, void* d_ws, size_t ws_size,
                              hipStream_t stream)
{
    (void)n_in; (void)out_size; (void)ws_size;
    const float* kp  = (const float*)d_in[0];
    const float* pos = (const float*)d_in[1];
    const int*   ei  = (const int*)d_in[2];
    const int N = in_sizes[1] / 3;   // 16384
    const int E = in_sizes[2] / 2;   // 262144
    const int R = in_sizes[0] / 4;   // N * 32 point rows
    const int* esrc = ei;
    const int* edst = ei + E;

    // params: tree-leaves order (dict keys sorted): cls, init, layers(f,g,h), loc
    int p = 3;
    auto nf = [&]() { return (const float*)d_in[p++]; };
    const float *cw1 = nf(), *cb1 = nf(), *cw2 = nf(), *cb2 = nf();
    const float *iw[4], *ib[4];
    for (int t = 0; t < 4; ++t) { iw[t] = nf(); ib[t] = nf(); }
    const float *fw1[3], *fb1[3], *fw2[3], *fb2[3];
    const float *gw1[3], *gb1[3], *gw2[3], *gb2[3];
    const float *hw1[3], *hb1[3], *hw2[3], *hb2[3];
    for (int t = 0; t < 3; ++t) {
        fw1[t] = nf(); fb1[t] = nf(); fw2[t] = nf(); fb2[t] = nf();
        gw1[t] = nf(); gb1[t] = nf(); gw2[t] = nf(); gb2[t] = nf();
        hw1[t] = nf(); hb1[t] = nf(); hw2[t] = nf(); hb2[t] = nf();
    }
    const float *lw[3][3], *lb[3][3];
    for (int c = 0; c < 3; ++c)
        for (int l = 0; l < 3; ++l) { lw[c][l] = nf(); lb[c][l] = nf(); }

    // workspace carving (deterministic)
    char* base = (char*)d_ws;
    size_t off = 0;
    auto carve = [&](size_t bytes) -> void* {
        void* r = base + off;
        off = (off + bytes + 255) & ~(size_t)255;
        return r;
    };
    float*    s    = (float*)carve((size_t)N * 300 * 4);
    u16*      sbm  = (u16*)carve((size_t)N * 320 * 2);   // bf16 mirror, padded
    float*    dxb  = (float*)carve((size_t)N * 16 * 4);
    unsigned* agg  = (unsigned*)carve((size_t)N * 300 * 4);
    float*    Wf32 = (float*)carve((size_t)303 * 300 * 4);
    float*    bf32 = (float*)carve(300 * 4);

    struct PW { u16* W; float* b; int Kpad, Otiles; };
    auto mk = [&](int I, int O) -> PW {
        PW q;
        q.Kpad = (I + 31) & ~31;
        int Opad = (O + 15) & ~15;
        q.Otiles = Opad / 16;
        q.W = (u16*)carve((size_t)2 * q.Kpad * Opad);
        q.b = (float*)carve((size_t)4 * Opad);
        return q;
    };
    auto prep = [&](const PW& q, const float* w, const float* b, int I, int O, int perm) {
        dim3 g((q.Kpad + 127) / 128, q.Otiles * 16);
        k_prep<<<g, 128, 0, stream>>>(w, b, q.W, q.b, I, O, q.Kpad, perm);
    };

    PW pwI[4] = { mk(4, 32), mk(32, 64), mk(64, 128), mk(128, 300) };
    PW pwC1 = mk(300, 64), pwC2 = mk(64, 3);
    PW pwL[3][3];
    for (int c = 0; c < 3; ++c) {
        pwL[c][0] = mk(300, 64); pwL[c][1] = mk(64, 64); pwL[c][2] = mk(64, 7);
    }
    PW pwH[3], pwF[3], pwG[3];
    for (int t = 0; t < 3; ++t) { pwH[t] = mk(300, 3); pwF[t] = mk(303, 300); pwG[t] = mk(300, 300); }

    // plain (normed) weights
    const int idims[4] = {4, 32, 64, 128}, odims[4] = {32, 64, 128, 300};
    for (int t = 0; t < 4; ++t) prep(pwI[t], iw[t], ib[t], idims[t], odims[t], 0);
    prep(pwC1, cw1, cb1, 300, 64, 0);
    prep(pwC2, cw2, cb2, 64, 3, 0);
    for (int c = 0; c < 3; ++c) {
        prep(pwL[c][0], lw[c][0], lb[c][0], 300, 64, 0);
        prep(pwL[c][1], lw[c][1], lb[c][1], 64, 64, 0);
        prep(pwL[c][2], lw[c][2], lb[c][2], 64, 7, 0);
    }
    // fused lin2 weights (stream-serial scratch reuse is safe)
    for (int t = 0; t < 3; ++t) {
        k_fuse<<<dim3(1, 301), 64, 0, stream>>>(hw1[t], hb1[t], hw2[t], hb2[t], Wf32, bf32, 300, 64, 3);
        prep(pwH[t], Wf32, bf32, 300, 3, 0);
        k_fuse<<<dim3((300 + 63) / 64, 304), 64, 0, stream>>>(fw1[t], fb1[t], fw2[t], fb2[t], Wf32, bf32, 303, 300, 300);
        prep(pwF[t], Wf32, bf32, 303, 300, 1);   // edge feature permutation
        k_fuse<<<dim3((300 + 63) / 64, 301), 64, 0, stream>>>(gw1[t], gb1[t], gw2[t], gb2[t], Wf32, bf32, 300, 300, 300);
        prep(pwG[t], Wf32, bf32, 300, 300, 0);
    }

    (void)hipMemsetAsync(s, 0, (size_t)N * 300 * 4, stream);
    k_init<<<dim3((R + 15) / 16), 32, 0, stream>>>(
        kp, pwI[0].W, pwI[0].b, pwI[1].W, pwI[1].b,
        pwI[2].W, pwI[2].b, pwI[3].W, pwI[3].b, s);
    k_s2bf<<<dim3((N * 320 + 255) / 256), 256, 0, stream>>>(s, sbm, N);

    for (int t = 0; t < 3; ++t) {
        k_agg_init<<<dim3((N * 300 + 255) / 256), 256, 0, stream>>>(agg, N * 300);
        k_dx<<<dim3((N + 15) / 16), 32, 0, stream>>>(sbm, pwH[t].W, pwH[t].b, dxb);
        k_edge<<<dim3((E + 15) / 16), 32, 0, stream>>>(sbm, pos, dxb, esrc, edst,
                                                       pwF[t].W, pwF[t].b, agg);
        k_update<<<dim3((N + 15) / 16), 32, 0, stream>>>(s, sbm, agg, pwG[t].W, pwG[t].b);
    }

    HeadW hw;
    hw.W[0] = pwC1.W; hw.B[0] = pwC1.b;
    hw.W[1] = pwC2.W; hw.B[1] = pwC2.b;
    for (int c = 0; c < 3; ++c)
        for (int l = 0; l < 3; ++l) { hw.W[2 + 3 * c + l] = pwL[c][l].W; hw.B[2 + 3 * c + l] = pwL[c][l].b; }
    k_heads<<<dim3((N + 15) / 16), 32, 0, stream>>>(sbm, hw, (float*)d_out, N);
}
